// MultiHeadedAttention_67413806678719
// MI455X (gfx1250) — matrix operand order repaired and checked by reading, not compiled, not run on hardware
//
#include <hip/hip_runtime.h>
#include <hip/hip_bf16.h>

// ---------------------------------------------------------------------------
// Problem constants (reference: B=8, L=1024, H=8, DK=DV=64, DM=512)
// ---------------------------------------------------------------------------
constexpr int kB  = 8;
constexpr int kL  = 1024;
constexpr int kH  = 8;
constexpr int kD  = 64;    // DK == DV
constexpr int kDM = 512;
constexpr int kBH = kB * kH;
constexpr int kNR = 65;    // REL_VOCAB = 2*32+1
constexpr int kM  = kB * kL;   // 8192 rows for projections

typedef __attribute__((ext_vector_type(16))) _Float16 v16h;
typedef __attribute__((ext_vector_type(2)))  _Float16 h2;
typedef __attribute__((ext_vector_type(8)))  float    v8f;
typedef __attribute__((ext_vector_type(4)))  unsigned int u32x4;
typedef __attribute__((ext_vector_type(8)))  unsigned int u32x8;

// ---------------------------------------------------------------------------
// WMMA fragment layouts (wave32, v_wmma_f32_16x16x32_f16), k order used here, not the ISA table order:
//  A (16x32): lane = m + 16g ; elem i -> k = i + 16g
//  B (32x16): lane = n + 16g ; elem i -> k = i + 16g
//  C/D (16x16 f32): vgpr j, lane = n + 16g -> m = j + 8g
// LDS tiles are stored PRE-SWIZZLED in fragment order: a fragment occupies a
// 512-halfword region laid out [lane][16 elems]; a fragment load is one
// contiguous 32B read per lane (2x ds_load_b128).
// ---------------------------------------------------------------------------
static __device__ __forceinline__ int a_slot(int m, int k) {
  return ((m + ((k >> 4) << 4)) << 4) + (k & 15);
}

static __device__ __forceinline__ int b_slot(int k, int n) {
  return ((n + ((k >> 4) << 4)) << 4) + (k & 15);
}

static __device__ __forceinline__ v16h load_frag(const _Float16* sm) {
  return *(const v16h*)(sm + (threadIdx.x & 31) * 16);
}

#define WMMA_F16(a, b, c) \
  __builtin_amdgcn_wmma_f32_16x16x32_f16(false, (a), false, (b), (short)0, (c), false, false)

// ---------------------------------------------------------------------------
// Tensor Data Mover: async 2D tile load into LDS (CDNA5 ISA).
//   dsz_code: 0=1B 1=2B 2=4B 3=8B ; dim0 = row length (elements) ;
//   rows = tile_dim1 ; stride = row stride in elements.
// Issued once per executing wave; tracked by TENSORcnt.
// ---------------------------------------------------------------------------
static __device__ __forceinline__ void tdm_load_2d(unsigned lds_off, const void* gptr,
                                                   unsigned dsz_code, unsigned dim0,
                                                   unsigned rows, unsigned stride) {
  unsigned long long ga = (unsigned long long)gptr;
  u32x4 g0;
  g0[0] = 1u;                                       // count=1, user mode
  g0[1] = lds_off;                                  // lds_addr
  g0[2] = (unsigned)ga;                             // global_addr[31:0]
  g0[3] = ((unsigned)(ga >> 32) & 0x01FFFFFFu)      // global_addr[56:32]
          | 0x80000000u;                            // type=2 ("image")
  u32x8 g1;
  g1[0] = dsz_code << 16;     // workgroup_mask=0, data_size
  g1[1] = dim0 << 16;         // tensor_dim0[15:0]      (bits 79:48)
  g1[2] = rows << 16;         // tensor_dim1[15:0]      (bits 111:80)
  g1[3] = dim0 << 16;         // tile_dim0              (bits 127:112)
  g1[4] = rows;               // tile_dim1              (bits 143:128)
  g1[5] = stride;             // tensor_dim0_stride     (bits 207:160)
  g1[6] = 0u;
  g1[7] = 0u;
  asm volatile("tensor_load_to_lds %0, %1" :: "s"(g0), "s"(g1) : "memory");
}

static __device__ __forceinline__ unsigned lds_offset_of(const void* p) {
  return (unsigned)__builtin_amdgcn_readfirstlane((int)(unsigned long long)(uintptr_t)p);
}

// ---------------------------------------------------------------------------
// Kernel 1: projection GEMM  out_f16(head-major) = (A_f32 @ W_f32 + bias)*scale
// TDM streams A/W k-tiles into LDS landing zones one step ahead; threads do an
// LDS->LDS convert/swizzle repack, then 2 WMMAs per wave per k-step.
// ---------------------------------------------------------------------------
__global__ __launch_bounds__(256)
void proj_gemm(const float* __restrict__ A, const float* __restrict__ W,
               const float* __restrict__ bias, _Float16* __restrict__ out,
               float scale) {
  __shared__ __align__(16) float    tdA[64 * 32];   // TDM landing (f32)
  __shared__ __align__(16) float    tdW[32 * 64];   // TDM landing (f32)
  __shared__ __align__(32) _Float16 sA[2048];       // 4 qb-regions x 512
  __shared__ __align__(32) _Float16 sB[2048];       // 4 n-regions  x 512

  const int tid  = threadIdx.x;
  const int wave = tid >> 5;
  const int lane = tid & 31;
  const int nlo  = lane & 15, g = lane >> 4;
  const int m0 = blockIdx.x * 64;
  const int n0 = blockIdx.y * 64;
  const int qb = wave & 3;
  const int nb = wave >> 2;

  const unsigned aoff = lds_offset_of(&tdA[0]);
  const unsigned woff = lds_offset_of(&tdW[0]);
  if (wave == 0) {
    tdm_load_2d(aoff, A + (size_t)m0 * kDM, 2, 32, 64, kDM);
    tdm_load_2d(woff, W + n0, 2, 64, 32, kDM);
  }

  v8f acc0 = {}, acc1 = {};

  for (int k0 = 0; k0 < kDM; k0 += 32) {
    if (wave == 0) __builtin_amdgcn_s_wait_tensorcnt(0);
    __syncthreads();

    // repack: A 64x32 f32 -> swizzled f16 (pair-merged)
#pragma unroll
    for (int r = 0; r < 4; ++r) {
      int pidx = tid + r * 256;
      int row = pidx >> 4;
      int kp  = (pidx & 15) * 2;
      float2 f = *(const float2*)(tdA + row * 32 + kp);
      h2 hv; hv[0] = (_Float16)f.x; hv[1] = (_Float16)f.y;
      *(h2*)(sA + ((row >> 4) << 9) + a_slot(row & 15, kp)) = hv;
    }
    // repack: W 32x64 f32 -> swizzled f16 (float4 LDS reads)
#pragma unroll
    for (int r = 0; r < 2; ++r) {
      int qidx = tid + r * 256;
      int row = qidx >> 4;            // k
      int cq  = (qidx & 15) * 4;      // n quad
      float4 f = *(const float4*)(tdW + row * 64 + cq);
      float fv[4] = {f.x, f.y, f.z, f.w};
#pragma unroll
      for (int u = 0; u < 4; ++u) {
        int col = cq + u;
        sB[((col >> 4) << 9) + b_slot(row, col & 15)] = (_Float16)fv[u];
      }
    }
    __syncthreads();

    // landing buffers are consumed: stream the next k-tile while we compute
    if (wave == 0 && k0 + 32 < kDM) {
      tdm_load_2d(aoff, A + (size_t)m0 * kDM + k0 + 32, 2, 32, 64, kDM);
      tdm_load_2d(woff, W + (size_t)(k0 + 32) * kDM + n0, 2, 64, 32, kDM);
    }

    v16h a  = load_frag(sA + (qb << 9));
    v16h b0 = load_frag(sB + ((nb * 2) << 9));
    v16h b1 = load_frag(sB + ((nb * 2 + 1) << 9));
    acc0 = WMMA_F16(a, b0, acc0);
    acc1 = WMMA_F16(a, b1, acc1);
    __syncthreads();
  }

#pragma unroll
  for (int j = 0; j < 8; ++j) {
    int m  = m0 + 16 * qb + j + 8 * g;
    int b_ = m >> 10, l = m & (kL - 1);
    {
      int n = n0 + 32 * nb + nlo;
      float v = (acc0[j] + bias[n]) * scale;
      out[((size_t)(b_ * kH + (n >> 6)) * kL + l) * kD + (n & 63)] = (_Float16)v;
    }
    {
      int n = n0 + 32 * nb + 16 + nlo;
      float v = (acc1[j] + bias[n]) * scale;
      out[((size_t)(b_ * kH + (n >> 6)) * kL + l) * kD + (n & 63)] = (_Float16)v;
    }
  }
}

// ---------------------------------------------------------------------------
// Kernel 2: qrel[bh][q][r] = sum_d q_f16[bh][q][d] * rel_emb_k[r][d]  (tiny)
// ---------------------------------------------------------------------------
__global__ __launch_bounds__(256)
void qrel_kernel(const _Float16* __restrict__ qbf,
                 const float* __restrict__ relk, float* __restrict__ qrel) {
  long idx = (long)blockIdx.x * 256 + threadIdx.x;
  const long total = (long)kBH * kL * kNR;
  if (idx >= total) return;
  int r = (int)(idx % kNR);
  long ql = idx / kNR;
  const _Float16* qp = qbf + ql * kD;
  const float* rp = relk + (size_t)r * kD;
  float acc = 0.f;
#pragma unroll
  for (int d = 0; d < kD; ++d) acc += (float)qp[d] * rp[d];
  qrel[idx] = acc;
}

// ---------------------------------------------------------------------------
// Kernel 3: fused flash attention, rel-bucket trick + tree bias.
// grid = (L/64, B*H); block = 256 = 8 waves. K/V tiles arrive via TDM one
// iteration ahead; waves repack LDS->LDS into fragment order during overlap.
// ---------------------------------------------------------------------------
__global__ __launch_bounds__(256)
void attn_kernel(const _Float16* __restrict__ qbf,
                 const _Float16* __restrict__ kbf,
                 const _Float16* __restrict__ vbf,
                 const float* __restrict__ qrel_g,        // [BH][L][65]
                 const float* __restrict__ relv_g,        // [65][64]
                 const float* __restrict__ tree_emb,      // [64][8]
                 const int* __restrict__ rel_matrix,      // [B][L][L]
                 const float* __restrict__ rel_mask,      // [B][L][L]
                 const unsigned char* __restrict__ maskp, // [B][L][L] bool
                 _Float16* __restrict__ ctx) {            // [B][L][512]
  __shared__ __align__(32) _Float16 qs[4096];     // 4 qb x 2 d-half x 512 (A)
  __shared__ __align__(32) _Float16 ksf[2048];    // 2 hb x 2 d-half x 512 (B^T)
  __shared__ __align__(32) _Float16 vsf[2048];    // 4 d-blk x 512 (B)
  __shared__ __align__(32) _Float16 pfr[2048];    // 4 qb x 512 (A, P matrix)
  __shared__ __align__(32) _Float16 tdK[2048];    // TDM landing, row-major 32x64
  __shared__ __align__(32) _Float16 tdV[2048];
  __shared__ float Ssm[64 * 36];                  // raw scores [64 q][32 key]
  __shared__ float qrel_s[64 * 66];
  __shared__ float pmass[64 * 66];
  __shared__ float relv_s[kNR * kD];
  __shared__ float treeh[64];
  __shared__ float mrow[64], lrow[64], crow[64];

  const int tid  = threadIdx.x;
  const int wave = tid >> 5;
  const int lane = tid & 31;
  const int nlo  = lane & 15, g = lane >> 4;
  const int qb = wave & 3;
  const int hb = wave >> 2;               // 0..1
  const int q0 = blockIdx.x * 64;
  const int bh = blockIdx.y;
  const int b_ = bh >> 3, h_ = bh & 7;
  const size_t kvbase = (size_t)bh * kL;

  const unsigned koff = lds_offset_of(&tdK[0]);
  const unsigned voff = lds_offset_of(&tdV[0]);

  // kick off TDM for tile 0 before the (long) prologue staging
  if (wave == 0) {
    tdm_load_2d(koff, kbf + kvbase * kD, 1, 64, 32, 64);
    tdm_load_2d(voff, vbf + kvbase * kD, 1, 64, 32, 64);
  }

  // ---- prologue: stage Q (swizzled), qrel rows, rel_emb_v, tree column ----
  const _Float16* qbase = qbf + (kvbase + q0) * kD;
#pragma unroll
  for (int r = 0; r < 8; ++r) {
    int pidx = tid + r * 256;
    int row = pidx >> 5;
    int dp  = (pidx & 31) * 2;
    h2 hv = *(const h2*)(qbase + (size_t)row * kD + dp);
    *(h2*)(qs + ((row >> 4) << 10) + (((dp >> 5) & 1) << 9) + a_slot(row & 15, dp & 31)) = hv;
  }
  for (int idx = tid; idx < 64 * kNR; idx += 256) {
    int row = idx / kNR, col = idx % kNR;
    qrel_s[row * 66 + col] = qrel_g[((size_t)bh * kL + q0 + row) * kNR + col];
  }
  for (int idx = tid; idx < kNR * kD; idx += 256) relv_s[idx] = relv_g[idx];
  if (tid < 64) {
    treeh[tid] = tree_emb[tid * kH + h_];
    mrow[tid] = -3.0e38f;
    lrow[tid] = 0.f;
  }
  for (int idx = tid; idx < 64 * 66; idx += 256) pmass[idx] = 0.f;
  __syncthreads();

  v8f acc0 = {}, acc1 = {};

  for (int kt = 0; kt < kL / 32; ++kt) {
    const int kkey0 = kt * 32;
    // ---- wait for TDM tile, make visible to all waves ----
    if (wave == 0) __builtin_amdgcn_s_wait_tensorcnt(0);
    __syncthreads();

    // ---- repack row-major TDM tiles into fragment-ordered ksf/vsf ----
#pragma unroll
    for (int r = 0; r < 4; ++r) {
      int pidx = tid + r * 256;
      int row = pidx >> 5;               // key 0..31
      int dp  = (pidx & 31) * 2;         // d
      h2 kv = *(const h2*)(tdK + row * 64 + dp);
      *(h2*)(ksf + ((row >> 4) << 10) + (((dp >> 5) & 1) << 9) +
             (((row & 15) + (((dp >> 4) & 1) << 4)) << 4) + (dp & 15)) = kv;
      h2 vv = *(const h2*)(tdV + row * 64 + dp);
      vsf[((dp >> 4) << 9) + (((dp & 15) + ((row >> 4) << 4)) << 4) + (row & 15)] = vv[0];
      int dp1 = dp + 1;
      vsf[((dp1 >> 4) << 9) + (((dp1 & 15) + ((row >> 4) << 4)) << 4) + (row & 15)] = vv[1];
    }
    __syncthreads();

    // ---- issue TDM for next tile (landing buffers are free now) ----
    if (wave == 0 && kt + 1 < kL / 32) {
      tdm_load_2d(koff, kbf + (kvbase + kkey0 + 32) * kD, 1, 64, 32, 64);
      tdm_load_2d(voff, vbf + (kvbase + kkey0 + 32) * kD, 1, 64, 32, 64);
    }

    // ---- S = q @ K^T : wave computes 16x16 tile at (16*qb, 16*hb) ----
    v8f s = {};
    {
      v16h a0 = load_frag(qs + (qb << 10));             // d 0..31
      v16h b0 = load_frag(ksf + (hb << 10));
      s = WMMA_F16(a0, b0, s);
      v16h a1 = load_frag(qs + (qb << 10) + 512);       // d 32..63
      v16h b1 = load_frag(ksf + (hb << 10) + 512);
      s = WMMA_F16(a1, b1, s);
    }
    // per-element: + bucketed qrel + tree bias, apply mask, spill raw scores
#pragma unroll
    for (int j = 0; j < 8; ++j) {
      int ml   = 16 * qb + j + 8 * g;
      int kcol = 16 * hb + nlo;
      int qg   = q0 + ml;
      int keyg = kkey0 + kcol;
      int dd = keyg - qg;
      dd = dd < -32 ? -32 : (dd > 32 ? 32 : dd);
      float sv = s[j] + qrel_s[ml * 66 + dd + 32];
      size_t e = ((size_t)b_ * kL + qg) * kL + keyg;
      sv += treeh[rel_matrix[e]] * rel_mask[e];
      if (maskp[e]) sv = -1.0e18f;
      Ssm[ml * 36 + kcol] = sv;
    }
    __syncthreads();

    // ---- online softmax: one thread per query row; P written straight into
    //      the fragment-ordered f16 buffer (pair stores) ----
    if (tid < 64) {
      float m_old = mrow[tid];
      float mx = m_old;
#pragma unroll
      for (int c = 0; c < 32; ++c) mx = fmaxf(mx, Ssm[tid * 36 + c]);
      float corr = __expf(m_old - mx);
      for (int r2 = 0; r2 < kNR; ++r2) pmass[tid * 66 + r2] *= corr;
      float sum = 0.f;
      int qg = q0 + tid;
      _Float16* prow = pfr + ((tid >> 4) << 9);
#pragma unroll
      for (int c = 0; c < 32; c += 2) {
        float p0 = __expf(Ssm[tid * 36 + c] - mx);
        float p1 = __expf(Ssm[tid * 36 + c + 1] - mx);
        sum += p0 + p1;
        h2 hv; hv[0] = (_Float16)p0; hv[1] = (_Float16)p1;
        *(h2*)(prow + a_slot(tid & 15, c)) = hv;
        int d0 = (kkey0 + c) - qg;
        d0 = d0 < -32 ? -32 : (d0 > 32 ? 32 : d0);
        pmass[tid * 66 + d0 + 32] += p0;
        int d1 = (kkey0 + c + 1) - qg;
        d1 = d1 < -32 ? -32 : (d1 > 32 ? 32 : d1);
        pmass[tid * 66 + d1 + 32] += p1;
      }
      lrow[tid] = lrow[tid] * corr + sum;
      mrow[tid] = mx;
      crow[tid] = corr;
    }
    __syncthreads();

    // ---- O += P @ V (rescale accumulators first) ----
#pragma unroll
    for (int j = 0; j < 8; ++j) {
      float c = crow[16 * qb + j + 8 * g];
      acc0[j] *= c;
      acc1[j] *= c;
    }
    v16h pa  = load_frag(pfr + (qb << 9));
    v16h vb0 = load_frag(vsf + ((hb * 2) << 9));
    v16h vb1 = load_frag(vsf + ((hb * 2 + 1) << 9));
    acc0 = WMMA_F16(pa, vb0, acc0);
    acc1 = WMMA_F16(pa, vb1, acc1);
    __syncthreads();
  }

  // ---- epilogue: (O + pmass @ rel_emb_v) / l, store ctx f16 row-major ----
#pragma unroll
  for (int j = 0; j < 8; ++j) {
    int ml = 16 * qb + j + 8 * g;
    float inv = 1.f / lrow[ml];
    int qg = q0 + ml;
    {
      int d = 32 * hb + nlo;
      float rv = 0.f;
      for (int rr = 0; rr < kNR; ++rr) rv += pmass[ml * 66 + rr] * relv_s[rr * kD + d];
      ctx[((size_t)b_ * kL + qg) * kDM + h_ * kD + d] = (_Float16)((acc0[j] + rv) * inv);
    }
    {
      int d = 32 * hb + 16 + nlo;
      float rv = 0.f;
      for (int rr = 0; rr < kNR; ++rr) rv += pmass[ml * 66 + rr] * relv_s[rr * kD + d];
      ctx[((size_t)b_ * kL + qg) * kDM + h_ * kD + d] = (_Float16)((acc1[j] + rv) * inv);
    }
  }
}

// ---------------------------------------------------------------------------
// Kernel 4: output GEMM  out_f32 = ctx_f16 @ Wo_f32 + bo  (TDM-pipelined)
// ---------------------------------------------------------------------------
__global__ __launch_bounds__(256)
void out_gemm(const _Float16* __restrict__ A, const float* __restrict__ W,
              const float* __restrict__ bias, float* __restrict__ out) {
  __shared__ __align__(16) _Float16 tdAh[64 * 32];  // TDM landing (f16)
  __shared__ __align__(16) float    tdW[32 * 64];   // TDM landing (f32)
  __shared__ __align__(32) _Float16 sA[2048];
  __shared__ __align__(32) _Float16 sB[2048];

  const int tid  = threadIdx.x;
  const int wave = tid >> 5;
  const int lane = tid & 31;
  const int nlo  = lane & 15, g = lane >> 4;
  const int m0 = blockIdx.x * 64;
  const int n0 = blockIdx.y * 64;
  const int qb = wave & 3;
  const int nb = wave >> 2;

  const unsigned aoff = lds_offset_of(&tdAh[0]);
  const unsigned woff = lds_offset_of(&tdW[0]);
  if (wave == 0) {
    tdm_load_2d(aoff, A + (size_t)m0 * kDM, 1, 32, 64, kDM);
    tdm_load_2d(woff, W + n0, 2, 64, 32, kDM);
  }

  v8f acc0 = {}, acc1 = {};

  for (int k0 = 0; k0 < kDM; k0 += 32) {
    if (wave == 0) __builtin_amdgcn_s_wait_tensorcnt(0);
    __syncthreads();

#pragma unroll
    for (int r = 0; r < 4; ++r) {
      int pidx = tid + r * 256;
      int row = pidx >> 4;
      int kp  = (pidx & 15) * 2;
      h2 hv = *(const h2*)(tdAh + row * 32 + kp);
      *(h2*)(sA + ((row >> 4) << 9) + a_slot(row & 15, kp)) = hv;
    }
#pragma unroll
    for (int r = 0; r < 2; ++r) {
      int qidx = tid + r * 256;
      int row = qidx >> 4;
      int cq  = (qidx & 15) * 4;
      float4 f = *(const float4*)(tdW + row * 64 + cq);
      float fv[4] = {f.x, f.y, f.z, f.w};
#pragma unroll
      for (int u = 0; u < 4; ++u) {
        int col = cq + u;
        sB[((col >> 4) << 9) + b_slot(row, col & 15)] = (_Float16)fv[u];
      }
    }
    __syncthreads();

    if (wave == 0 && k0 + 32 < kDM) {
      tdm_load_2d(aoff, A + (size_t)m0 * kDM + k0 + 32, 1, 32, 64, kDM);
      tdm_load_2d(woff, W + (size_t)(k0 + 32) * kDM + n0, 2, 64, 32, kDM);
    }

    v16h a  = load_frag(sA + (qb << 9));
    v16h b0 = load_frag(sB + ((nb * 2) << 9));
    v16h b1 = load_frag(sB + ((nb * 2 + 1) << 9));
    acc0 = WMMA_F16(a, b0, acc0);
    acc1 = WMMA_F16(a, b1, acc1);
    __syncthreads();
  }

#pragma unroll
  for (int j = 0; j < 8; ++j) {
    int m = m0 + 16 * qb + j + 8 * g;
    out[(size_t)m * kDM + n0 + 32 * nb + nlo] = acc0[j] + bias[n0 + 32 * nb + nlo];
    out[(size_t)m * kDM + n0 + 32 * nb + 16 + nlo] = acc1[j] + bias[n0 + 32 * nb + 16 + nlo];
  }
}

// ---------------------------------------------------------------------------
// Host-side launcher
// ---------------------------------------------------------------------------
extern "C" void kernel_launch(void* const* d_in, const int* in_sizes, int n_in,
                              void* d_out, int out_size, void* d_ws, size_t ws_size,
                              hipStream_t stream) {
  const float* key            = (const float*)d_in[0];
  const float* value          = (const float*)d_in[1];
  const float* query          = (const float*)d_in[2];
  const unsigned char* maskp  = (const unsigned char*)d_in[3];
  const int*   rel_matrix     = (const int*)d_in[4];
  const float* rel_mask       = (const float*)d_in[5];
  const float* Wk = (const float*)d_in[6];
  const float* bk = (const float*)d_in[7];
  const float* Wq = (const float*)d_in[8];
  const float* bq = (const float*)d_in[9];
  const float* Wv = (const float*)d_in[10];
  const float* bv = (const float*)d_in[11];
  const float* Wo = (const float*)d_in[12];
  const float* bo = (const float*)d_in[13];
  const float* rel_emb_k = (const float*)d_in[14];
  const float* rel_emb_v = (const float*)d_in[15];
  const float* tree_emb  = (const float*)d_in[16];
  float* out = (float*)d_out;

  char* ws = (char*)d_ws;
  size_t off = 0;
  _Float16* qbf = (_Float16*)(ws + off); off += (size_t)kBH * kL * kD * sizeof(_Float16);
  _Float16* kbf = (_Float16*)(ws + off); off += (size_t)kBH * kL * kD * sizeof(_Float16);
  _Float16* vbf = (_Float16*)(ws + off); off += (size_t)kBH * kL * kD * sizeof(_Float16);
  float*    qrel = (float*)(ws + off);   off += (size_t)kBH * kL * kNR * sizeof(float);
  _Float16* ctx = (_Float16*)(ws + off); off += (size_t)kB * kL * kDM * sizeof(_Float16);
  (void)off; (void)ws_size; (void)in_sizes; (void)n_in; (void)out_size;

  dim3 gproj(kM / 64, kDM / 64);   // 128 x 8
  const float qscale = 0.125f;     // 1/sqrt(DK)
  proj_gemm<<<gproj, 256, 0, stream>>>(query, Wq, bq, qbf, qscale);
  proj_gemm<<<gproj, 256, 0, stream>>>(key,   Wk, bk, kbf, 1.0f);
  proj_gemm<<<gproj, 256, 0, stream>>>(value, Wv, bv, vbf, 1.0f);

  long qrel_total = (long)kBH * kL * kNR;
  qrel_kernel<<<(unsigned)((qrel_total + 255) / 256), 256, 0, stream>>>(qbf, rel_emb_k, qrel);

  attn_kernel<<<dim3(kL / 64, kBH), 256, 0, stream>>>(
      qbf, kbf, vbf, qrel, rel_emb_v, tree_emb, rel_matrix, rel_mask, maskp, ctx);

  out_gemm<<<gproj, 256, 0, stream>>>(ctx, Wo, bo, out);
}
